// HPB_36455682408774
// MI455X (gfx1250) — matrix operand order repaired and checked by reading, not compiled, not run on hardware
//
#include <hip/hip_runtime.h>
#include <math.h>

// ---------------------------------------------------------------------------
// HPB forward for MI455X (gfx1250, wave32, WMMA).
// GEMMs: v_wmma_f32_16x16x32_bf16, fp32 accumulate, 128x64 block tile,
// 32x32 per wave (4 WMMAs/step), double-buffered LDS with async-LDS fills.
// ---------------------------------------------------------------------------

typedef __attribute__((ext_vector_type(16))) __bf16 v16bf;
typedef __attribute__((ext_vector_type(8)))  float  v8f;

union V16U { float4 f4[2]; v16bf v; __bf16 h[16]; };
union B8U  { float4 f;     __bf16 h[8]; };

#define WMMA_BF16(a, b, c) \
  __builtin_amdgcn_wmma_f32_16x16x32_bf16(false, (a), false, (b), (short)0, (c), false, false)

// Async global->LDS 16B copy per lane (gfx1250 GLOBAL_LOAD_ASYNC_TO_LDS_B128).
#define ASYNC_COPY_B128(lds_u32v, gaddr_u64) \
  asm volatile("global_load_async_to_lds_b128 %0, %1, off" \
               :: "v"(lds_u32v), "v"(gaddr_u64) : "memory")
#define WAIT_ASYNC() asm volatile("s_wait_asynccnt 0" ::: "memory")

__device__ __forceinline__ unsigned lds_u32(const void* p) {
  // DS addressing is relative to the wave's LDS allocation; low 32 bits of the
  // generic shared-aperture pointer hold the LDS byte offset.
  return (unsigned)(unsigned long long)(size_t)p;
}
__device__ __forceinline__ __bf16 f2bf(float v) { return (__bf16)v; }
__device__ __forceinline__ float  gelu_exact(float x) {
  return 0.5f * x * (1.f + erff(x * 0.70710678118654752f));
}

// ---------------------------------------------------------------------------
// fp32 -> bf16 conversion
// ---------------------------------------------------------------------------
__global__ void cvt_bf16_kernel(const float* __restrict__ S, __bf16* __restrict__ D, int n) {
  int i = blockIdx.x * 256 + threadIdx.x;
  if (i < n) D[i] = f2bf(S[i]);
}

// ---------------------------------------------------------------------------
// Channel LayerNorm over C=256 (per pixel), writes bf16
// ---------------------------------------------------------------------------
__global__ __launch_bounds__(256) void chan_ln_kernel(const float* __restrict__ X,
                                                      const float* __restrict__ G,
                                                      const float* __restrict__ Bc,
                                                      __bf16* __restrict__ XN) {
  int pix = blockIdx.x * 256 + threadIdx.x;   // 0..65535
  int b = pix >> 12, p = pix & 4095;
  const float* xb = X + ((long)b << 20) + p;
  float s = 0.f, q = 0.f;
  for (int c = 0; c < 256; ++c) { float v = xb[(long)c << 12]; s += v; q += v * v; }
  float m = s * (1.f / 256.f);
  float var = q * (1.f / 256.f) - m * m;
  float rstd = rsqrtf(var + 1e-5f);
  __bf16* xo = XN + ((long)b << 20) + p;
  for (int c = 0; c < 256; ++c)
    xo[(long)c << 12] = f2bf((xb[(long)c << 12] - m) * rstd * G[c] + Bc[c]);
}

// ---------------------------------------------------------------------------
// WMMA bf16 GEMM:  Y[b,o,p] = sum_c W[o,c] * X[b,c,p]  (+bias, +resid)
// Block tile 128(M) x 64(N), K step 32, double-buffered LDS.
// 8 waves as 4(M) x 2(N); each wave: 32x32 output = 4 accumulators.
//   As [m][k] row-major, filled by async-LDS copies (ASYNCcnt).
//   Bt [n][k] transposed, filled by float4 load + scatter ds_store_b16.
// ---------------------------------------------------------------------------
template <bool HB, bool HR, bool OF, bool OB>
__global__ __launch_bounds__(256) void gemm_bf16_kernel(
    const __bf16* __restrict__ W, const __bf16* __restrict__ X,
    const float* __restrict__ bias, const float* __restrict__ resid,
    float* __restrict__ outF, __bf16* __restrict__ outB,
    int O, int C, long sxb, long syb) {
  __shared__ __align__(16) __bf16 As[2][128][40];  // [buf][m][k]
  __shared__ __align__(16) __bf16 Bt[2][64][40];   // [buf][n][k]
  const int tid = threadIdx.x;
  const int lane = tid & 31, wave = tid >> 5;
  const int l15 = lane & 15, hi = lane >> 4;
  const int wm = wave >> 1, wn = wave & 1;
  const int n0 = blockIdx.x * 64;
  const int m0 = blockIdx.y * 128;
  const int bz = blockIdx.z;
  const long P = 4096;
  const __bf16* Xb = X + (long)bz * sxb;

  v8f acc[2][2] = {{{}, {}}, {{}, {}}};

  // A tile: 128x32 = 512 16B chunks; 2 per thread
  const int ar0 = tid >> 2, ac0 = (tid & 3) * 8;
  const int ar1 = (tid + 256) >> 2, ac1 = ((tid + 256) & 3) * 8;
  // B tile: 32x64; one float4 per thread, scatter-transposed
  const int bkk = tid >> 3, bns = (tid & 7) * 8;

  const int nk = C >> 5;
#define FILL_A(buf, k0)                                                          \
  do {                                                                           \
    ASYNC_COPY_B128(lds_u32(&As[buf][ar0][ac0]),                                 \
        (unsigned long long)(size_t)(W + (long)(m0 + ar0) * C + (k0) + ac0));    \
    ASYNC_COPY_B128(lds_u32(&As[buf][ar1][ac1]),                                 \
        (unsigned long long)(size_t)(W + (long)(m0 + ar1) * C + (k0) + ac1));    \
  } while (0)
#define FILL_B(buf, k0)                                                          \
  do {                                                                           \
    B8U ub_;                                                                     \
    ub_.f = *(const float4*)(Xb + (long)((k0) + bkk) * P + n0 + bns);            \
    _Pragma("unroll") for (int i_ = 0; i_ < 8; ++i_)                             \
        Bt[buf][bns + i_][bkk] = ub_.h[i_];                                      \
  } while (0)

  FILL_A(0, 0);
  FILL_B(0, 0);

  for (int k = 0; k < nk; ++k) {
    const int cur = k & 1;
    WAIT_ASYNC();
    __syncthreads();  // buf[cur] ready on all waves; buf[cur^1] reads done
    if (k + 1 < nk) {
      FILL_A(cur ^ 1, (k + 1) << 5);
      FILL_B(cur ^ 1, (k + 1) << 5);
    }
    // A fragments (two 16-row tiles), each = two contiguous 16B runs per lane
    V16U a0, a1;
    {
      const __bf16* r = &As[cur][wm * 32 + l15][0];
      a0.f4[0] = *(const float4*)(r + hi * 16);
      a0.f4[1] = *(const float4*)(r + 8 + hi * 16);
      const __bf16* r2 = &As[cur][wm * 32 + 16 + l15][0];
      a1.f4[0] = *(const float4*)(r2 + hi * 16);
      a1.f4[1] = *(const float4*)(r2 + 8 + hi * 16);
    }
    // B fragments (two 16-col tiles) from transposed tile
    V16U b0, b1;
    {
      const __bf16* r = &Bt[cur][wn * 32 + l15][hi * 16];
      b0.f4[0] = *(const float4*)(r);
      b0.f4[1] = *(const float4*)(r + 8);
      const __bf16* r2 = &Bt[cur][wn * 32 + 16 + l15][hi * 16];
      b1.f4[0] = *(const float4*)(r2);
      b1.f4[1] = *(const float4*)(r2 + 8);
    }
    acc[0][0] = WMMA_BF16(a0.v, b0.v, acc[0][0]);
    acc[0][1] = WMMA_BF16(a0.v, b1.v, acc[0][1]);
    acc[1][0] = WMMA_BF16(a1.v, b0.v, acc[1][0]);
    acc[1][1] = WMMA_BF16(a1.v, b1.v, acc[1][1]);
  }
#undef FILL_A
#undef FILL_B

  // Epilogue: C/D layout lane=N, VGPR r -> M=r (lo half) / r+8 (hi half)
#pragma unroll
  for (int mi = 0; mi < 2; ++mi) {
#pragma unroll
    for (int r = 0; r < 8; ++r) {
      int m = m0 + wm * 32 + mi * 16 + r + hi * 8;
      float bv = HB ? bias[m] : 0.f;
#pragma unroll
      for (int ni = 0; ni < 2; ++ni) {
        int n = n0 + wn * 32 + ni * 16 + l15;
        float v = acc[mi][ni][r] + bv;
        if (HR) v += resid[(long)bz * O * P + (long)m * P + n];
        long yi = (long)bz * syb + (long)m * P + n;
        if (OF) outF[yi] = v;
        if (OB) outB[yi] = f2bf(v);
      }
    }
  }
}

// ---------------------------------------------------------------------------
// L2 norm along width (rows of 64) for q,k channels of qkv, inplace
// ---------------------------------------------------------------------------
__global__ void l2norm_kernel(float* __restrict__ qkv) {
  long r = (long)blockIdx.x * 256 + threadIdx.x;  // 16*512*64 rows
  int b = (int)(r >> 15);
  int rem = (int)(r & 32767);
  int ch = rem >> 6;
  int hh = rem & 63;
  float* p = qkv + ((long)b * 768 + ch) * 4096 + hh * 64;
  float s = 0.f;
  for (int w = 0; w < 64; ++w) s += p[w] * p[w];
  float inv = 1.f / fmaxf(sqrtf(s), 1e-12f);
  for (int w = 0; w < 64; ++w) p[w] *= inv;
}

// ---------------------------------------------------------------------------
// Top-8 row / top-8 col selection + gather -> k_sel/v_sel (BH,64,32) bf16
// ---------------------------------------------------------------------------
__global__ __launch_bounds__(256) void select_kernel(const float* __restrict__ qkv,
                                                     __bf16* __restrict__ ksel,
                                                     __bf16* __restrict__ vsel) {
  __shared__ float probe[32];
  __shared__ float part[256];
  __shared__ float score[64];
  __shared__ int idxh[8];
  __shared__ int idxw[8];
  const int tid = threadIdx.x;
  const int bh = blockIdx.x;
  const int b = bh >> 3, head = bh & 7;
  const long P = 4096;
  const float* q = qkv + ((long)b * 768 + head * 32) * P;
  const float* k = qkv + ((long)b * 768 + 256 + head * 32) * P;
  const float* v = qkv + ((long)b * 768 + 512 + head * 32) * P;

  {
    int d = tid >> 3, seg = tid & 7;
    float s = 0.f;
    const float* qd = q + (long)d * P + seg * 512;
    for (int i = 0; i < 512; ++i) s += fabsf(qd[i]);
    part[tid] = s;
  }
  __syncthreads();
  if (tid < 32) {
    float s = 0.f;
    for (int i = 0; i < 8; ++i) s += part[tid * 8 + i];
    probe[tid] = s;
  }
  __syncthreads();

  {
    int h = tid & 63, dg = tid >> 6;
    float s = 0.f;
    for (int d = dg * 8; d < dg * 8 + 8; ++d) {
      const float* kr = k + (long)d * P + h * 64;
      float rs = 0.f;
      for (int w = 0; w < 64; ++w) rs += fabsf(kr[w]);
      s += probe[d] * rs;
    }
    part[tid] = s;
  }
  __syncthreads();
  if (tid < 64) score[tid] = part[tid] + part[64 + tid] + part[128 + tid] + part[192 + tid];
  __syncthreads();
  if (tid == 0) {
    for (int i = 0; i < 8; ++i) {
      int am = 0; float bm = -1e30f;
      for (int h = 0; h < 64; ++h) if (score[h] > bm) { bm = score[h]; am = h; }
      idxh[i] = am; score[am] = -1e30f;
    }
  }
  __syncthreads();

  {
    int w = tid & 63, dg = tid >> 6;
    float s = 0.f;
    for (int d = dg * 8; d < dg * 8 + 8; ++d) {
      float cs = 0.f;
      for (int i = 0; i < 8; ++i) cs += fabsf(k[(long)d * P + idxh[i] * 64 + w]);
      s += probe[d] * cs;
    }
    part[tid] = s;
  }
  __syncthreads();
  if (tid < 64) score[tid] = part[tid] + part[64 + tid] + part[128 + tid] + part[192 + tid];
  __syncthreads();
  if (tid == 0) {
    for (int i = 0; i < 8; ++i) {
      int am = 0; float bm = -1e30f;
      for (int w = 0; w < 64; ++w) if (score[w] > bm) { bm = score[w]; am = w; }
      idxw[i] = am; score[am] = -1e30f;
    }
  }
  __syncthreads();

  for (int e = tid; e < 2048; e += 256) {
    int j = e >> 5, d = e & 31;
    int hh = idxh[j >> 3], ww = idxw[j & 7];
    long sp = (long)d * P + hh * 64 + ww;
    ksel[(long)bh * 2048 + e] = f2bf(k[sp]);
    vsel[(long)bh * 2048 + e] = f2bf(v[sp]);
  }
}

// ---------------------------------------------------------------------------
// Attention: one block per (b,head) x 128 queries; each wave 16 queries.
//   Ks [n][d]: async-LDS copy (k_sel layout is already fragment-friendly).
//   Vt [d][n]: scatter-transposed so P*V B-fragments are contiguous.
// ---------------------------------------------------------------------------
__global__ __launch_bounds__(256) void attn_kernel(const float* __restrict__ qkv,
                                                   const __bf16* __restrict__ ksel,
                                                   const __bf16* __restrict__ vsel,
                                                   __bf16* __restrict__ rawout) {
  __shared__ __align__(16) __bf16 Qs[128][40];     // [m][d]
  __shared__ __align__(16) __bf16 Ks[64][40];      // [n][d]
  __shared__ __align__(16) __bf16 Vt[32][72];      // [d][n]
  __shared__ __align__(16) __bf16 Ps[8][16][72];   // per-wave probs [m][n]
  const int tid = threadIdx.x;
  const int lane = tid & 31, wave = tid >> 5;
  const int l15 = lane & 15, hi = lane >> 4;
  const int bh = blockIdx.y;
  const int b = bh >> 3, head = bh & 7;
  const int p0 = blockIdx.x * 128;
  const long P = 4096;
  const float* qb = qkv + ((long)b * 768 + head * 32) * P;

  {
    int n = tid >> 2, seg = (tid & 3) * 8;
    ASYNC_COPY_B128(lds_u32(&Ks[n][seg]),
        (unsigned long long)(size_t)(ksel + (long)bh * 2048 + n * 32 + seg));
  }
  {
    int n = tid >> 2, seg = (tid & 3) * 8;
    B8U uv;
    uv.f = *(const float4*)(vsel + (long)bh * 2048 + n * 32 + seg);
#pragma unroll
    for (int i = 0; i < 8; ++i) Vt[seg + i][n] = uv.h[i];
  }
  for (int e = tid; e < 128 * 32; e += 256) {
    int m = e & 127, d = e >> 7;
    Qs[m][d] = f2bf(qb[(long)d * P + p0 + m]);
  }
  WAIT_ASYNC();
  __syncthreads();

  // sim = Q * Ksel^T (16x64 per wave)
  V16U aq;
  {
    const __bf16* row = &Qs[wave * 16 + l15][0];
    aq.f4[0] = *(const float4*)(row + hi * 16);
    aq.f4[1] = *(const float4*)(row + 8 + hi * 16);
  }
  v8f s[4] = {{}, {}, {}, {}};
#pragma unroll
  for (int nt = 0; nt < 4; ++nt) {
    V16U bk;
    const __bf16* rK = &Ks[nt * 16 + l15][hi * 16];
    bk.f4[0] = *(const float4*)(rK);
    bk.f4[1] = *(const float4*)(rK + 8);
    s[nt] = WMMA_BF16(aq.v, bk.v, s[nt]);
  }

  // softmax per query row
#pragma unroll
  for (int r = 0; r < 8; ++r) {
    float mx = -1e30f;
#pragma unroll
    for (int nt = 0; nt < 4; ++nt) mx = fmaxf(mx, s[nt][r]);
#pragma unroll
    for (int msk = 1; msk <= 8; msk <<= 1) mx = fmaxf(mx, __shfl_xor(mx, msk, 32));
    float e0[4], sum = 0.f;
#pragma unroll
    for (int nt = 0; nt < 4; ++nt) { e0[nt] = __expf(s[nt][r] - mx); sum += e0[nt]; }
#pragma unroll
    for (int msk = 1; msk <= 8; msk <<= 1) sum += __shfl_xor(sum, msk, 32);
    float inv = 1.f / sum;
    int rr = r + hi * 8;
#pragma unroll
    for (int nt = 0; nt < 4; ++nt) Ps[wave][rr][nt * 16 + l15] = f2bf(e0[nt] * inv);
  }
  // per-wave LDS RAW: DS ops are in-order within a wave

  // out = P * V  (16x64 * 64x32), K split into two 32-chunks
  v8f o0 = {}, o1 = {};
#pragma unroll
  for (int kc = 0; kc < 2; ++kc) {
    V16U ap;
    {
      const __bf16* row = &Ps[wave][l15][kc * 32];
      ap.f4[0] = *(const float4*)(row + hi * 16);
      ap.f4[1] = *(const float4*)(row + 8 + hi * 16);
    }
    V16U bv0, bv1;
    const __bf16* r0 = &Vt[l15][kc * 32 + hi * 16];
    bv0.f4[0] = *(const float4*)(r0);
    bv0.f4[1] = *(const float4*)(r0 + 8);
    const __bf16* r1 = &Vt[16 + l15][kc * 32 + hi * 16];
    bv1.f4[0] = *(const float4*)(r1);
    bv1.f4[1] = *(const float4*)(r1 + 8);
    o0 = WMMA_BF16(ap.v, bv0.v, o0);
    o1 = WMMA_BF16(ap.v, bv1.v, o1);
  }

#pragma unroll
  for (int r = 0; r < 8; ++r) {
    int p = p0 + wave * 16 + r + hi * 8;
    rawout[((long)b * 256 + head * 32 + l15) * P + p] = f2bf(o0[r]);
    rawout[((long)b * 256 + head * 32 + 16 + l15) * P + p] = f2bf(o1[r]);
  }
}

// ---------------------------------------------------------------------------
// Depthwise 3x3 conv, SAME padding.
// ---------------------------------------------------------------------------
__global__ void dwconv3_kernel(const float* __restrict__ X, const float* __restrict__ Wd,
                               const float* __restrict__ Bd, float* __restrict__ outF,
                               __bf16* __restrict__ outB, int C, int outCh, int chOff) {
  long e = (long)blockIdx.x * 256 + threadIdx.x;
  int p = (int)(e & 4095);
  long bcx = e >> 12;
  int c = (int)(bcx % C);
  int b = (int)(bcx / C);
  int h = p >> 6, w = p & 63;
  const float* Xc = X + (((long)b * C + c) << 12);
  const float* wk = Wd + c * 9;
  float s = Bd[c];
#pragma unroll
  for (int kh = -1; kh <= 1; ++kh) {
    int hh = h + kh;
    if (hh < 0 || hh > 63) continue;
#pragma unroll
    for (int kw = -1; kw <= 1; ++kw) {
      int wwp = w + kw;
      if (wwp < 0 || wwp > 63) continue;
      s += wk[(kh + 1) * 3 + (kw + 1)] * Xc[hh * 64 + wwp];
    }
  }
  long oi = ((long)b * outCh + chOff + c) * 4096 + p;
  if (outF) outF[oi] = s;
  if (outB) outB[oi] = f2bf(s);
}

// ---------------------------------------------------------------------------
// Instance norm variants (one block per (b,channel))
// ---------------------------------------------------------------------------
__device__ __forceinline__ void in_stats(const float* __restrict__ Y, long base, int tid,
                                         float* ssum, float* ssq, float& m, float& rstd) {
  float s = 0.f, q = 0.f;
  for (int i = tid; i < 4096; i += 256) { float v = Y[base + i]; s += v; q += v * v; }
  ssum[tid] = s; ssq[tid] = q; __syncthreads();
  for (int st = 128; st > 0; st >>= 1) {
    if (tid < st) { ssum[tid] += ssum[tid + st]; ssq[tid] += ssq[tid + st]; }
    __syncthreads();
  }
  m = ssum[0] * (1.f / 4096.f);
  float var = ssq[0] * (1.f / 4096.f) - m * m;
  rstd = rsqrtf(var + 1e-5f);
}

__global__ __launch_bounds__(256) void in_gelu_kernel(const float* __restrict__ Y,
                                                      float* __restrict__ H) {
  __shared__ float ssum[256], ssq[256];
  const int tid = threadIdx.x;
  const long base = (long)blockIdx.x * 4096;
  float m, rstd;
  in_stats(Y, base, tid, ssum, ssq, m, rstd);
  for (int i = tid; i < 4096; i += 256)
    H[base + i] = gelu_exact((Y[base + i] - m) * rstd);
}

__global__ __launch_bounds__(256) void in_gelu_add_kernel(const float* __restrict__ T,
                                                          const float* __restrict__ H,
                                                          __bf16* __restrict__ H2) {
  __shared__ float ssum[256], ssq[256];
  const int tid = threadIdx.x;
  const long base = (long)blockIdx.x * 4096;
  float m, rstd;
  in_stats(T, base, tid, ssum, ssq, m, rstd);
  for (int i = tid; i < 4096; i += 256)
    H2[base + i] = f2bf(H[base + i] + gelu_exact((T[base + i] - m) * rstd));
}

__global__ __launch_bounds__(256) void in_final_kernel(const float* __restrict__ Y,
                                                       float* __restrict__ OUT) {
  __shared__ float ssum[256], ssq[256];
  const int tid = threadIdx.x;
  const long base = (long)blockIdx.x * 4096;
  float m, rstd;
  in_stats(Y, base, tid, ssum, ssq, m, rstd);
  for (int i = tid; i < 4096; i += 256)
    OUT[base + i] = (Y[base + i] - m) * rstd;
}

// ---------------------------------------------------------------------------
// Host orchestration
// ---------------------------------------------------------------------------
extern "C" void kernel_launch(void* const* d_in, const int* in_sizes, int n_in,
                              void* d_out, int out_size, void* d_ws, size_t ws_size,
                              hipStream_t stream) {
  (void)in_sizes; (void)n_in; (void)out_size; (void)ws_size;
  const float* x      = (const float*)d_in[0];
  const float* g      = (const float*)d_in[1];
  const float* bc     = (const float*)d_in[2];
  const float* w_qkv  = (const float*)d_in[3];
  const float* w_out  = (const float*)d_in[4];
  const float* b_out  = (const float*)d_in[5];
  const float* w_dw   = (const float*)d_in[6];
  const float* b_dw   = (const float*)d_in[7];
  const float* w_comb = (const float*)d_in[8];
  const float* b_comb = (const float*)d_in[9];
  const float* w_ff1  = (const float*)d_in[10];
  const float* b_ff1  = (const float*)d_in[11];
  const float* w_ffdw = (const float*)d_in[12];
  const float* b_ffdw = (const float*)d_in[13];
  const float* w_ff2  = (const float*)d_in[14];
  const float* b_ff2  = (const float*)d_in[15];
  float* out = (float*)d_out;
  char* ws = (char*)d_ws;

  const long P = 4096;
  const int B = 16;

  size_t off = 0;
  auto alloc = [&](size_t bytes) { size_t o = off; off += (bytes + 255) & ~(size_t)255; return o; };
  size_t o_wqkv = alloc((size_t)768 * 256 * 2);
  size_t o_wout = alloc((size_t)256 * 256 * 2);
  size_t o_wcmb = alloc((size_t)256 * 512 * 2);
  size_t o_wff1 = alloc((size_t)1024 * 256 * 2);
  size_t o_wff2 = alloc((size_t)256 * 1024 * 2);
  size_t o_xnbf = alloc((size_t)B * 256 * P * 2);
  size_t o_qkv  = alloc((size_t)B * 768 * P * 4);    // f32 (reused as h2 bf16)
  size_t o_ksel = alloc((size_t)128 * 64 * 32 * 2);
  size_t o_vsel = alloc((size_t)128 * 64 * 32 * 2);
  size_t o_raw  = alloc((size_t)B * 256 * P * 2);
  size_t o_cat  = alloc((size_t)B * 512 * P * 2);    // (reused as y2 f32)
  size_t o_attn = alloc((size_t)B * 256 * P * 2);
  size_t o_y1   = alloc((size_t)B * 1024 * P * 4);   // (reused as t)
  size_t o_h    = alloc((size_t)B * 1024 * P * 4);

  __bf16* wqkv_bf = (__bf16*)(ws + o_wqkv);
  __bf16* wout_bf = (__bf16*)(ws + o_wout);
  __bf16* wcmb_bf = (__bf16*)(ws + o_wcmb);
  __bf16* wff1_bf = (__bf16*)(ws + o_wff1);
  __bf16* wff2_bf = (__bf16*)(ws + o_wff2);
  __bf16* xnbf    = (__bf16*)(ws + o_xnbf);
  float*  qkv     = (float*)(ws + o_qkv);
  __bf16* ksel    = (__bf16*)(ws + o_ksel);
  __bf16* vsel    = (__bf16*)(ws + o_vsel);
  __bf16* rawbf   = (__bf16*)(ws + o_raw);
  __bf16* catbf   = (__bf16*)(ws + o_cat);
  __bf16* attnbf  = (__bf16*)(ws + o_attn);
  float*  y1      = (float*)(ws + o_y1);
  float*  hbuf    = (float*)(ws + o_h);
  float*  tbuf    = y1;
  __bf16* h2bf    = (__bf16*)(ws + o_qkv);
  float*  y2      = (float*)(ws + o_cat);

  cvt_bf16_kernel<<<(768 * 256 + 255) / 256, 256, 0, stream>>>(w_qkv, wqkv_bf, 768 * 256);
  cvt_bf16_kernel<<<(256 * 256 + 255) / 256, 256, 0, stream>>>(w_out, wout_bf, 256 * 256);
  cvt_bf16_kernel<<<(256 * 512 + 255) / 256, 256, 0, stream>>>(w_comb, wcmb_bf, 256 * 512);
  cvt_bf16_kernel<<<(1024 * 256 + 255) / 256, 256, 0, stream>>>(w_ff1, wff1_bf, 1024 * 256);
  cvt_bf16_kernel<<<(256 * 1024 + 255) / 256, 256, 0, stream>>>(w_ff2, wff2_bf, 256 * 1024);

  chan_ln_kernel<<<256, 256, 0, stream>>>(x, g, bc, xnbf);

  // qkv: O=768 -> grid.y = 768/128 = 6
  gemm_bf16_kernel<false, false, true, false><<<dim3(64, 6, 16), 256, 0, stream>>>(
      wqkv_bf, xnbf, nullptr, nullptr, qkv, nullptr, 768, 256, 256 * P, 768 * P);

  l2norm_kernel<<<(16 * 512 * 64) / 256, 256, 0, stream>>>(qkv);
  select_kernel<<<128, 256, 0, stream>>>(qkv, ksel, vsel);
  attn_kernel<<<dim3(32, 128), 256, 0, stream>>>(qkv, ksel, vsel, rawbf);

  // w_out: O=256 -> grid.y = 2
  gemm_bf16_kernel<true, false, false, true><<<dim3(64, 2, 16), 256, 0, stream>>>(
      wout_bf, rawbf, b_out, nullptr, nullptr, catbf, 256, 256, 256 * P, 512 * P);

  dwconv3_kernel<<<(16 * 256 * 4096) / 256, 256, 0, stream>>>(
      x, w_dw, b_dw, nullptr, catbf, 256, 512, 256);

  gemm_bf16_kernel<true, true, false, true><<<dim3(64, 2, 16), 256, 0, stream>>>(
      wcmb_bf, catbf, b_comb, x, nullptr, attnbf, 256, 512, 512 * P, 256 * P);

  // ff1: O=1024 -> grid.y = 8
  gemm_bf16_kernel<true, false, true, false><<<dim3(64, 8, 16), 256, 0, stream>>>(
      wff1_bf, attnbf, b_ff1, nullptr, y1, nullptr, 1024, 256, 256 * P, 1024 * P);

  in_gelu_kernel<<<16 * 1024, 256, 0, stream>>>(y1, hbuf);

  dwconv3_kernel<<<(16 * 1024 * 4096) / 256, 256, 0, stream>>>(
      hbuf, w_ffdw, b_ffdw, tbuf, nullptr, 1024, 1024, 0);

  in_gelu_add_kernel<<<16 * 1024, 256, 0, stream>>>(tbuf, hbuf, h2bf);

  gemm_bf16_kernel<true, false, true, false><<<dim3(64, 2, 16), 256, 0, stream>>>(
      wff2_bf, h2bf, b_ff2, nullptr, y2, nullptr, 256, 1024, 1024 * P, 256 * P);

  in_final_kernel<<<16 * 256, 256, 0, stream>>>(y2, out);
}
